// ForwardLSTM_60799557042438
// MI455X (gfx1250) — matrix operand order repaired and checked by reading, not compiled, not run on hardware
//
#include <hip/hip_runtime.h>
#include <math.h>

#define B_  256
#define T_  512
#define H_  512
#define G4_ 2048   // 4*H

typedef __attribute__((ext_vector_type(16))) __bf16 v16bf;
typedef __attribute__((ext_vector_type(8)))  float  v8f;

// ---------------------------------------------------------------------------
// Pack Wh (f32 [H][4H] row-major) into bf16 WMMA B-fragment tiles.
// B 32x16 bf16 layout used here (contiguous k = 16*half + e, not the ISA table order): VGPR j, lanes 0-15: N=lane, K=2j,2j+1;
// lanes 16-31: N=lane-16, K=16+2j,16+2j+1.  Tile order: (ntile, ktile).
// Per-lane fragment = 16 consecutive bf16 (32B) -> two b128 loads.
// ---------------------------------------------------------------------------
__global__ __launch_bounds__(256) void pack_wh_kernel(const float* __restrict__ Wh,
                                                      __bf16* __restrict__ Wp) {
  const int tid = blockIdx.x * blockDim.x + threadIdx.x;   // 0 .. H*4H-1
  const int k = tid / G4_;
  const int n = tid % G4_;
  const int ntile = n >> 4, nn = n & 15;
  const int ktile = k >> 5, kk = k & 31;
  const int lane = nn + ((kk >> 4) << 4);
  const int v = (kk & 15) >> 1;
  const int e = (v << 1) + (kk & 1);
  Wp[(((ntile * 16 + ktile) * 32 + lane) << 4) + e] = (__bf16)Wh[tid];
}

__global__ __launch_bounds__(256) void init_state_kernel(float* __restrict__ c,
                                                         __bf16* __restrict__ hp) {
  const int tid = blockIdx.x * blockDim.x + threadIdx.x;   // 0 .. B*H-1
  c[tid]  = 0.0f;
  hp[tid] = (__bf16)0.0f;
}

// ---------------------------------------------------------------------------
// Per-row LayerNorm + ReLU + (H x 3) projection using one wave32, shfl-only.
// ---------------------------------------------------------------------------
__device__ __forceinline__ void ln_row(const float* __restrict__ row,
                                       const float* __restrict__ scale,
                                       const float* __restrict__ lbias,
                                       const float* __restrict__ Wd,
                                       const float* __restrict__ bd,
                                       float* __restrict__ out3, int lane) {
  float v[16];
  float s = 0.0f, sq = 0.0f;
#pragma unroll
  for (int i = 0; i < 16; ++i) {
    const float f = row[i * 32 + lane];
    v[i] = f;
    s += f;
    sq += f * f;
  }
#pragma unroll
  for (int o = 16; o > 0; o >>= 1) {
    s  += __shfl_xor(s,  o, 32);
    sq += __shfl_xor(sq, o, 32);
  }
  const float mu  = s * (1.0f / H_);
  const float var = sq * (1.0f / H_) - mu * mu;
  const float rs  = rsqrtf(var + 1e-6f);
  float p0 = 0.0f, p1 = 0.0f, p2 = 0.0f;
#pragma unroll
  for (int i = 0; i < 16; ++i) {
    const int j = i * 32 + lane;
    const float y = fmaxf((v[i] - mu) * rs * scale[j] + lbias[j], 0.0f);
    p0 += y * Wd[j * 3 + 0];
    p1 += y * Wd[j * 3 + 1];
    p2 += y * Wd[j * 3 + 2];
  }
#pragma unroll
  for (int o = 16; o > 0; o >>= 1) {
    p0 += __shfl_xor(p0, o, 32);
    p1 += __shfl_xor(p1, o, 32);
    p2 += __shfl_xor(p2, o, 32);
  }
  if (lane == 0) {
    out3[0] = p0 + bd[0];
    out3[1] = p1 + bd[1];
    out3[2] = p2 + bd[2];
  }
}

// ---------------------------------------------------------------------------
// One LSTM timestep, fully fused, PLUS LayerNorm+projection of timestep t-1.
//   Blocks 0..511  : Z = h_prev@Wh (bf16 WMMA, f32 acc) + b + x_t@Wi,
//                    gates -> c,h update -> h_cur (f32) + bf16 A-frag packed
//   Blocks 512..543: LN+ReLU+proj of h from t-1 (reads other h buffer)
// block = 128 threads = 4 waves.
// ---------------------------------------------------------------------------
__global__ __launch_bounds__(128) void lstm_step_kernel(
    const float* __restrict__ x, const float* __restrict__ Wi,
    const float* __restrict__ bias, const __bf16* __restrict__ Wp,
    __bf16* __restrict__ hp, float* __restrict__ c_state,
    float* __restrict__ h_cur, const float* __restrict__ h_prev,
    const float* __restrict__ lns, const float* __restrict__ lnb,
    const float* __restrict__ Wd, const float* __restrict__ bd,
    float* __restrict__ out, int t) {
  __shared__ float gbuf[4][16][17];

  const int bid  = blockIdx.x;
  const int wv   = threadIdx.x >> 5;
  const int lane = threadIdx.x & 31;

  if (bid < 512) {
    const int nt   = bid & 31;        // 16-wide column tile inside a gate
    const int mt   = bid >> 5;        // 16-high batch row tile
    const int half = lane >> 4;
    const int nn   = lane & 15;

    // ---- GEMM: 16x16 tile of gate wv, K = H = 512 in 16 steps of 32 ----
    v8f acc = {};
    const int ntp = wv * 32 + nt;     // global 16-col tile 0..127
    const __bf16* ap = hp + (((mt  * 16) * 32 + lane) << 4);
    const __bf16* bp = Wp + (((ntp * 16) * 32 + lane) << 4);
#pragma unroll
    for (int kt = 0; kt < 16; ++kt) {
      v16bf a = *(const v16bf*)(ap + ((kt * 32) << 4));
      v16bf b = *(const v16bf*)(bp + ((kt * 32) << 4));
      acc = __builtin_amdgcn_wmma_f32_16x16x32_bf16(false, a, false, b,
                                                    (short)0, acc, false, false);
    }

    // ---- epilogue: + bias + x_t @ Wi (F_IN=3), activation, stash in LDS ----
    const int   gcol = wv * H_ + nt * 16 + nn;
    const float wi0  = Wi[gcol];
    const float wi1  = Wi[G4_ + gcol];
    const float wi2  = Wi[2 * G4_ + gcol];
    const float bb   = bias[gcol];
#pragma unroll
    for (int r = 0; r < 8; ++r) {
      const int    brow = mt * 16 + half * 8 + r;       // C/D layout: M = r + 8*half
      const float* xr   = x + ((size_t)brow * T_ + t) * 3;
      const float  z    = acc[r] + bb + xr[0] * wi0 + xr[1] * wi1 + xr[2] * wi2;
      const float  g    = (wv == 2) ? tanhf(z) : (1.0f / (1.0f + expf(-z)));
      gbuf[wv][half * 8 + r][nn] = g;
    }
    __syncthreads();

    // ---- cell/hidden update: 256 tile elements on 128 threads ----
#pragma unroll
    for (int it = 0; it < 2; ++it) {
      const int idx = threadIdx.x + it * 128;
      const int bl  = idx >> 4;
      const int jl  = idx & 15;
      const float ig = gbuf[0][bl][jl];
      const float fg = gbuf[1][bl][jl];
      const float gg = gbuf[2][bl][jl];
      const float og = gbuf[3][bl][jl];
      const int brow = mt * 16 + bl;
      const int jcol = nt * 16 + jl;
      const int off  = brow * H_ + jcol;
      const float cn = fg * c_state[off] + ig * gg;
      c_state[off]   = cn;
      const float hv = og * tanhf(cn);
      h_cur[off]     = hv;
      // scatter into bf16 A-fragment layout (16-bit A 16x32, ISA 7.12.2)
      const int ktile = jcol >> 5, kk = jcol & 31;
      const int lh = kk >> 4;
      const int vv = (kk & 15) >> 1;
      const int ee = (vv << 1) + (kk & 1);
      hp[(((mt * 16 + ktile) * 32 + (bl + 16 * lh)) << 4) + ee] = (__bf16)hv;
    }
  } else if (t > 0) {
    // ---- LN blocks: 32 blocks x 4 waves x 2 rows = 256 rows of timestep t-1
    const int q = bid - 512;            // 0..31
#pragma unroll
    for (int rr = 0; rr < 2; ++rr) {
      const int b = q * 8 + wv * 2 + rr;
      ln_row(h_prev + (size_t)b * H_, lns, lnb, Wd, bd,
             out + ((size_t)b * T_ + (t - 1)) * 3, lane);
    }
  }
}

// ---------------------------------------------------------------------------
// Standalone LN+proj for the final timestep (t = T-1).
// grid = 32 blocks x 128 threads; same wave-per-row scheme.
// ---------------------------------------------------------------------------
__global__ __launch_bounds__(128) void lnproj_final_kernel(
    const float* __restrict__ h_full, const float* __restrict__ lns,
    const float* __restrict__ lnb, const float* __restrict__ Wd,
    const float* __restrict__ bd, float* __restrict__ out, int t) {
  const int wv   = threadIdx.x >> 5;
  const int lane = threadIdx.x & 31;
#pragma unroll
  for (int rr = 0; rr < 2; ++rr) {
    const int b = blockIdx.x * 8 + wv * 2 + rr;
    ln_row(h_full + (size_t)b * H_, lns, lnb, Wd, bd,
           out + ((size_t)b * T_ + t) * 3, lane);
  }
}

// ---------------------------------------------------------------------------
extern "C" void kernel_launch(void* const* d_in, const int* in_sizes, int n_in,
                              void* d_out, int out_size, void* d_ws, size_t ws_size,
                              hipStream_t stream) {
  const float* x   = (const float*)d_in[0];
  const float* Wi  = (const float*)d_in[1];
  const float* Wh  = (const float*)d_in[2];
  const float* bv  = (const float*)d_in[3];
  const float* lns = (const float*)d_in[4];
  const float* lnb = (const float*)d_in[5];
  const float* Wd  = (const float*)d_in[6];
  const float* bd  = (const float*)d_in[7];
  float* out = (float*)d_out;

  // workspace carve-out (~3.75 MB)
  char* p = (char*)d_ws;
  __bf16* Wp      = (__bf16*)p;  p += (size_t)H_ * G4_ * sizeof(__bf16); // 2 MB
  __bf16* hp      = (__bf16*)p;  p += (size_t)B_ * H_  * sizeof(__bf16); // 256 KB
  float*  c_state = (float*)p;   p += (size_t)B_ * H_  * sizeof(float);  // 512 KB
  float*  h_buf0  = (float*)p;   p += (size_t)B_ * H_  * sizeof(float);  // 512 KB
  float*  h_buf1  = (float*)p;                                           // 512 KB

  pack_wh_kernel<<<(H_ * G4_) / 256, 256, 0, stream>>>(Wh, Wp);
  init_state_kernel<<<(B_ * H_) / 256, 256, 0, stream>>>(c_state, hp);

  for (int t = 0; t < T_; ++t) {
    float* h_cur        = (t & 1) ? h_buf1 : h_buf0;
    const float* h_prev = (t & 1) ? h_buf0 : h_buf1;
    lstm_step_kernel<<<544, 128, 0, stream>>>(x, Wi, bv, Wp, hp, c_state,
                                              h_cur, h_prev, lns, lnb, Wd, bd,
                                              out, t);
  }
  // LN+proj for the last timestep (T-1 is odd -> h_buf1 holds it)
  lnproj_final_kernel<<<32, 128, 0, stream>>>(h_buf1, lns, lnb, Wd, bd, out,
                                              T_ - 1);
}
